// BitLinearB158_53833120088620
// MI455X (gfx1250) — 8-bit matrix operand order repaired and checked by reading, compiled, not run on hardware
//
#include <hip/hip_runtime.h>
#include <stdint.h>

#define EPS 1e-5f
#define TM 128
#define TN 128
#define TK 64

typedef __attribute__((ext_vector_type(8))) int   v8i;
typedef __attribute__((ext_vector_type(4))) int   v4i;
typedef __attribute__((ext_vector_type(4))) float v4f;

#ifndef __has_builtin
#define __has_builtin(x) 0
#endif

#define GLOBAL_AS __attribute__((address_space(1)))
#define LDS_AS    __attribute__((address_space(3)))

__device__ __forceinline__ void async_copy_b128(const void* g, void* l) {
#if __has_builtin(__builtin_amdgcn_global_load_async_to_lds_b128)
  __builtin_amdgcn_global_load_async_to_lds_b128(
      (GLOBAL_AS v4i*)(void*)g, (LDS_AS v4i*)l, 0, 0);
#else
  *(v4i*)l = *(const v4i*)g;
#endif
}

__device__ __forceinline__ void wait_async_done() {
#if __has_builtin(__builtin_amdgcn_s_wait_asynccnt)
  __builtin_amdgcn_s_wait_asynccnt(0);
#else
  asm volatile("s_wait_asynccnt 0" ::: "memory");
#endif
}

__global__ void init_scalars(unsigned* gbits, float* bsum) {
  if (threadIdx.x == 0) { *gbits = 0u; *bsum = 0.0f; }
}

__global__ __launch_bounds__(256) void reduce_absmax(const float* __restrict__ x,
                                                     long n4, unsigned* gbits) {
  long gid = (long)blockIdx.x * blockDim.x + threadIdx.x;
  long stride = (long)gridDim.x * blockDim.x;
  const v4f* x4 = (const v4f*)x;
  float m = 0.0f;
  for (long i = gid; i < n4; i += stride) {
    v4f v = x4[i];
    m = fmaxf(m, fmaxf(fmaxf(fabsf(v.x), fabsf(v.y)), fmaxf(fabsf(v.z), fabsf(v.w))));
  }
#pragma unroll
  for (int o = 16; o > 0; o >>= 1) m = fmaxf(m, __shfl_xor(m, o, 32));
  __shared__ float sm[8];
  const int lane = threadIdx.x & 31, wv = threadIdx.x >> 5;
  if (lane == 0) sm[wv] = m;
  __syncthreads();
  if (threadIdx.x == 0) {
    float t = sm[0];
#pragma unroll
    for (int i = 1; i < 8; ++i) t = fmaxf(t, sm[i]);
    atomicMax(gbits, __float_as_uint(t));
  }
}

__global__ __launch_bounds__(256) void reduce_abssum(const float* __restrict__ w,
                                                     long n4, float* bsum) {
  long gid = (long)blockIdx.x * blockDim.x + threadIdx.x;
  long stride = (long)gridDim.x * blockDim.x;
  const v4f* w4 = (const v4f*)w;
  float s = 0.0f;
  for (long i = gid; i < n4; i += stride) {
    v4f v = w4[i];
    s += fabsf(v.x) + fabsf(v.y) + fabsf(v.z) + fabsf(v.w);
  }
#pragma unroll
  for (int o = 16; o > 0; o >>= 1) s += __shfl_xor(s, o, 32);
  __shared__ float sm[8];
  const int lane = threadIdx.x & 31, wv = threadIdx.x >> 5;
  if (lane == 0) sm[wv] = s;
  __syncthreads();
  if (threadIdx.x == 0) {
    float t = 0.0f;
#pragma unroll
    for (int i = 0; i < 8; ++i) t += sm[i];
    atomicAdd(bsum, t);
  }
}

__global__ __launch_bounds__(256) void quant_x_kernel(const float* __restrict__ x,
                                                      char* __restrict__ xq, long n4,
                                                      const unsigned* __restrict__ gbits) {
  long i = (long)blockIdx.x * blockDim.x + threadIdx.x;
  if (i >= n4) return;
  const float s = 128.0f / fmaxf(__uint_as_float(*gbits), EPS);
  v4f v = ((const v4f*)x)[i];
  int b0 = (int)fminf(fmaxf(rintf(v.x * s), -128.0f), 127.0f);
  int b1 = (int)fminf(fmaxf(rintf(v.y * s), -128.0f), 127.0f);
  int b2 = (int)fminf(fmaxf(rintf(v.z * s), -128.0f), 127.0f);
  int b3 = (int)fminf(fmaxf(rintf(v.w * s), -128.0f), 127.0f);
  ((int*)xq)[i] = (b0 & 255) | ((b1 & 255) << 8) | ((b2 & 255) << 16) | ((b3 & 255) << 24);
}

__global__ __launch_bounds__(256) void quant_w_kernel(const float* __restrict__ w,
                                                      char* __restrict__ wq, long n4,
                                                      const float* __restrict__ bsum,
                                                      float wcount_inv) {
  long i = (long)blockIdx.x * blockDim.x + threadIdx.x;
  if (i >= n4) return;
  const float inv = 1.0f / fmaxf(*bsum * wcount_inv, EPS);
  v4f v = ((const v4f*)w)[i];
  int b0 = (int)fminf(fmaxf(rintf(fabsf(v.x) * inv), -1.0f), 1.0f);
  int b1 = (int)fminf(fmaxf(rintf(fabsf(v.y) * inv), -1.0f), 1.0f);
  int b2 = (int)fminf(fmaxf(rintf(fabsf(v.z) * inv), -1.0f), 1.0f);
  int b3 = (int)fminf(fmaxf(rintf(fabsf(v.w) * inv), -1.0f), 1.0f);
  ((int*)wq)[i] = (b0 & 255) | ((b1 & 255) << 8) | ((b2 & 255) << 16) | ((b3 & 255) << 24);
}

__global__ __launch_bounds__(256) void gemm_wmma_iu8(
    const char* __restrict__ A, const char* __restrict__ B,
    float* __restrict__ out, int M, int N, int K,
    const unsigned* __restrict__ gbits,
    const float* __restrict__ bsum, float wcount_inv) {
  __shared__ char As[2][TM * TK];
  __shared__ char Bs[2][TN * TK];

  const int tid  = threadIdx.x;
  const int lane = tid & 31;
  const int wave = tid >> 5;
  const int m0 = blockIdx.y * TM;
  const int n0 = blockIdx.x * TN;
  const int wm = (wave & 1) * 64;
  const int wn = (wave >> 1) * 32;

  v8i acc[4][2] = {};

  const int c0 = tid * 2;
  auto prefetch = [&](int kt, int buf) {
    const long kk = (long)kt * TK;
#pragma unroll
    for (int i = 0; i < 2; ++i) {
      const int c = c0 + i;
      const int row = c >> 2;
      const int off = (c & 3) << 4;
      async_copy_b128(A + (long)(m0 + row) * K + kk + off, &As[buf][row * TK + off]);
      async_copy_b128(B + (long)(n0 + row) * K + kk + off, &Bs[buf][row * TK + off]);
    }
  };

  prefetch(0, 0);

  const int aRow = wm + (lane & 15);
  const int aOff = (lane >> 4) * 8;
  const int bRow = wn + (lane & 15);
  const int bOff = (lane >> 4) * 8;
  const int KT = K / TK;

  for (int kt = 0; kt < KT; ++kt) {
    const int cur = kt & 1;
    wait_async_done();
    __syncthreads();
    if (kt + 1 < KT) prefetch(kt + 1, cur ^ 1);

    v8i a[4], b[2];
#pragma unroll
    for (int mi = 0; mi < 4; ++mi) {
      const char* p = &As[cur][(aRow + mi * 16) * TK + aOff];
      union { unsigned long long d[4]; v8i v; } u;
      u.d[0] = *(const unsigned long long*)(p);
      u.d[1] = *(const unsigned long long*)(p + 16);
      u.d[2] = *(const unsigned long long*)(p + 32);
      u.d[3] = *(const unsigned long long*)(p + 48);
      a[mi] = u.v;
    }
#pragma unroll
    for (int ni = 0; ni < 2; ++ni) {
      const char* p = &Bs[cur][(bRow + ni * 16) * TK + bOff];
      union { unsigned long long d[4]; v8i v; } u;
      u.d[0] = *(const unsigned long long*)(p);
      u.d[1] = *(const unsigned long long*)(p + 16);
      u.d[2] = *(const unsigned long long*)(p + 32);
      u.d[3] = *(const unsigned long long*)(p + 48);
      b[ni] = u.v;
    }
#pragma unroll
    for (int mi = 0; mi < 4; ++mi)
#pragma unroll
      for (int ni = 0; ni < 2; ++ni)
        acc[mi][ni] = __builtin_amdgcn_wmma_i32_16x16x64_iu8(
            true, a[mi], true, b[ni], acc[mi][ni], false, false);
  }

  const float gamma = fmaxf(__uint_as_float(*gbits), EPS);
  const float beta  = fmaxf(*bsum * wcount_inv, EPS);
  const float scale = gamma * beta * (1.0f / 128.0f);

  const int cc = lane & 15, gg = lane >> 4;
#pragma unroll
  for (int mi = 0; mi < 4; ++mi)
#pragma unroll
    for (int ni = 0; ni < 2; ++ni) {
      float* o = out + (long)(m0 + wm + mi * 16 + 8 * gg) * N + (n0 + wn + ni * 16 + cc);
#pragma unroll
      for (int r = 0; r < 8; ++r)
        o[(long)r * N] = (float)acc[mi][ni][r] * scale;
    }
}

extern "C" void kernel_launch(void* const* d_in, const int* in_sizes, int n_in,
                              void* d_out, int out_size, void* d_ws, size_t ws_size,
                              hipStream_t stream) {
  const float* x = (const float*)d_in[0];
  const float* w = (const float*)d_in[1];
  float* out = (float*)d_out;

  const int K = 4096;
  const long nx = (long)in_sizes[0];
  const long nw = (long)in_sizes[1];
  const int M = (int)(nx / K);
  const int N = (int)(nw / K);
  const float wcount_inv = 1.0f / (float)nw;

  unsigned* gbits = (unsigned*)d_ws;
  float* bsum = (float*)((char*)d_ws + 8);
  char* xq = (char*)d_ws + 256;
  char* wq = xq + nx;

  init_scalars<<<1, 1, 0, stream>>>(gbits, bsum);
  reduce_absmax<<<2048, 256, 0, stream>>>(x, nx >> 2, gbits);
  reduce_abssum<<<2048, 256, 0, stream>>>(w, nw >> 2, bsum);
  quant_x_kernel<<<(int)(((nx >> 2) + 255) / 256), 256, 0, stream>>>(x, xq, nx >> 2, gbits);
  quant_w_kernel<<<(int)(((nw >> 2) + 255) / 256), 256, 0, stream>>>(w, wq, nw >> 2, bsum, wcount_inv);

  dim3 grid((unsigned)(N / TN), (unsigned)(M / TM));
  gemm_wmma_iu8<<<grid, 256, 0, stream>>>(xq, wq, out, M, N, K, gbits, bsum, wcount_inv);
}
